// informer_54468775248144
// MI455X (gfx1250) — compile-verified, matrix operand order checked by reading, not run on hardware
//
#include <hip/hip_runtime.h>
#include <cmath>

// ---------------------------------------------------------------------------
// Informer forward for MI455X (gfx1250, wave32, WMMA + TDM).
// Dense GEMMs: TENSOR_LOAD_TO_LDS stages double-buffered f32 tiles into LDS
// (descriptor per CDNA5 ISA ch.8, TENSORcnt + barrier sync), waves read
// fragments with ds_load_b128, convert to f16, and run
// v_wmma_f32_16x16x32_f16 with f32 accumulation.
// ---------------------------------------------------------------------------

#define DM   512     // d_model
#define NH   8       // heads
#define DH   64      // head dim
#define DFF  1024    // ffn dim
#define NB   4       // batch
#define SL   2048    // sequence length
#define KCOL 1536    // im2col K for distil conv (512*3)

typedef _Float16 f16_t;
typedef __attribute__((ext_vector_type(16))) _Float16 v16h;
typedef __attribute__((ext_vector_type(8)))  float    v8f;
typedef __attribute__((ext_vector_type(4)))  unsigned int v4u;
typedef __attribute__((ext_vector_type(8)))  int      v8i_t;
typedef __attribute__((ext_vector_type(4)))  int      v4i_t;

enum { ACT_NONE = 0, ACT_GELU = 1, ACT_RELU = 2, ACT_ELU = 3 };

// ------------------------------- device helpers ----------------------------

__device__ __forceinline__ float apply_act(float v, int act) {
    if (act == ACT_GELU) return 0.5f * v * (1.f + erff(v * 0.70710678118654752f));
    if (act == ACT_RELU) return fmaxf(v, 0.f);
    if (act == ACT_ELU)  return v > 0.f ? v : expm1f(v);
    return v;
}

// TDM 2D tile load: copies tileRows x 32 f32 elements (row stride ldElems)
// from global into contiguous LDS at byte offset ldsOff. rowsAvail clips the
// tensor in Y: rows beyond it are read as zero by the TDM (handles ragged N).
// Descriptor packing per CDNA5 ISA ch.8 (D# groups 0/1; groups 2/3 zero = 2D).
// This toolchain's builtin arity is 6:
//   (uint32x4 g0, int32x8 g1, int32x4 g2, int32x4 g3, int32x8 zero, i32 cpol)
__device__ __forceinline__ void tdm_load_tile(unsigned ldsOff, const float* gsrc,
                                              int tileRows, int rowsAvail, int ldElems)
{
    unsigned long long ga = (unsigned long long)(size_t)gsrc;
    v4u g0;
    g0[0] = 1u;                                                    // count=1 (user D#)
    g0[1] = ldsOff;                                                // lds_addr (bytes)
    g0[2] = (unsigned)ga;                                          // global_addr[31:0]
    g0[3] = (unsigned)((ga >> 32) & 0x01FFFFFFull) | (2u << 30);   // addr[56:32] | type=2

    unsigned td0 = 32u;                                            // tensor_dim0
    unsigned td1 = (unsigned)(rowsAvail < 0 ? 0 : rowsAvail);      // tensor_dim1 (clip)
    unsigned t0  = 32u;                                            // tile_dim0
    unsigned t1  = (unsigned)tileRows;                             // tile_dim1
    unsigned long long s0 = (unsigned long long)(unsigned)ldElems; // tensor_dim0_stride

    v8i_t g1;
    g1[0] = (int)(2u << 16);                                       // data_size = 4 bytes
    g1[1] = (int)((td0 & 0xFFFFu) << 16);                          // tensor_dim0 -> bits 63:48
    g1[2] = (int)((td0 >> 16) | ((td1 & 0xFFFFu) << 16));          // dim0 hi | dim1 lo
    g1[3] = (int)((td1 >> 16) | ((t0 & 0xFFFFu) << 16));           // dim1 hi | tile_dim0
    g1[4] = (int)(t1 & 0xFFFFu);                                   // tile_dim1 (tile_dim2=0)
    g1[5] = (int)(unsigned)(s0 & 0xFFFFFFFFull);                   // stride0 lo
    g1[6] = (int)(unsigned)((s0 >> 32) & 0xFFFFull);               // stride0 hi (dim1_stride=0)
    g1[7] = 0;

    v4i_t z4; z4[0] = 0; z4[1] = 0; z4[2] = 0; z4[3] = 0;
    v8i_t z8;
#pragma unroll
    for (int e = 0; e < 8; ++e) z8[e] = 0;
    __builtin_amdgcn_tensor_load_to_lds(g0, g1, z4, z4, z8, 0);
}

// CDNA5 16-bit 16x32 fragment lane layout (A rows / B cols, same mapping):
//   lanes 0-15 : index = base + lane,      K halves {0..7, 16..23}
//   lanes 16-31: index = base + lane - 16, K halves {8..15, 24..31}
// Source is an LDS tile laid out [rows][32] f32.
__device__ __forceinline__ v16h frag_from_lds(const float* __restrict__ T, int rbase)
{
    const int lane = threadIdx.x & 31;
    const float* p = T + (rbase + (lane & 15)) * 32 + ((lane >> 4) << 3);
    v16h f;
#pragma unroll
    for (int e = 0; e < 8; ++e) f[e] = (f16_t)p[e];
#pragma unroll
    for (int e = 0; e < 8; ++e) f[8 + e] = (f16_t)p[16 + e];
    return f;
}

// C/D layout: lanes 0-15 col=nt+lane rows mt+0..7 ; lanes 16-31 col=nt+lane-16 rows mt+8..15
__device__ __forceinline__ void store_tile(float* __restrict__ C,
                                           const float* __restrict__ bias,
                                           const float* __restrict__ R,
                                           v8f acc, int mt, int nt, int M, int N, int act) {
    const int lane  = threadIdx.x & 31;
    const int col   = nt + (lane & 15);
    const int rbase = mt + ((lane >> 4) << 3);
    if (col >= N) return;
    const float bs = bias ? bias[col] : 0.f;
#pragma unroll
    for (int r = 0; r < 8; ++r) {
        int row = rbase + r;
        if (row < M) {
            float v = acc[r] + bs;
            if (R) v += R[(size_t)row * N + col];
            C[(size_t)row * N + col] = apply_act(v, act);
        }
    }
}

// ------------------------------- GEMM (WMMA) -------------------------------
// C[M,N] = act(A[M,K] @ W[N,K]^T + bias + R). 8 waves/block, 32x32 per wave,
// 128x64 block tile. K multiple of 32 (true for all call sites). TDM stages
// double-buffered 128x32 (A) and 64x32 (B) f32 tiles into LDS.
__global__ __launch_bounds__(256) void gemm_wmma_kernel(
    const float* __restrict__ A, const float* __restrict__ W,
    const float* __restrict__ bias, const float* __restrict__ R,
    float* __restrict__ C, int M, int N, int K, int act)
{
    __shared__ __align__(16) float shA[2][128 * 32];   // 16 KB x2
    __shared__ __align__(16) float shB[2][64 * 32];    //  8 KB x2

    const int wave = threadIdx.x >> 5;
    const int mblk = blockIdx.y * 128;
    const int nblk = blockIdx.x * 64;
    const int m0 = mblk + (wave >> 1) * 32;
    const int n0 = nblk + (wave & 1) * 32;
    const int ml = (wave >> 1) * 32;      // local row base in A tile
    const int nl = (wave & 1) * 32;       // local row base in B tile

    v8f acc00 = {}; v8f acc01 = {}; v8f acc10 = {}; v8f acc11 = {};

    const int nk = K >> 5;
    if (wave == 0) {
        tdm_load_tile((unsigned)(size_t)&shA[0][0], A + (size_t)mblk * K, 128, M - mblk, K);
        tdm_load_tile((unsigned)(size_t)&shB[0][0], W + (size_t)nblk * K, 64,  N - nblk, K);
    }

    for (int ki = 0; ki < nk; ++ki) {
        __builtin_amdgcn_s_wait_tensorcnt((short)0);   // no-op for non-issuing waves
        __syncthreads();                               // tiles in `cur` visible to all
        const int cur = ki & 1;
        if (ki + 1 < nk && wave == 0) {                // prefetch next K-slice
            const int kb = (ki + 1) << 5;
            tdm_load_tile((unsigned)(size_t)&shA[cur ^ 1][0],
                          A + (size_t)mblk * K + kb, 128, M - mblk, K);
            tdm_load_tile((unsigned)(size_t)&shB[cur ^ 1][0],
                          W + (size_t)nblk * K + kb, 64, N - nblk, K);
        }
        const float* At = shA[cur];
        const float* Bt = shB[cur];
        v16h a0 = frag_from_lds(At, ml);
        v16h a1 = frag_from_lds(At, ml + 16);
        v16h b0 = frag_from_lds(Bt, nl);
        v16h b1 = frag_from_lds(Bt, nl + 16);
        acc00 = __builtin_amdgcn_wmma_f32_16x16x32_f16(false, a0, false, b0, (short)0, acc00, false, false);
        acc01 = __builtin_amdgcn_wmma_f32_16x16x32_f16(false, a0, false, b1, (short)0, acc01, false, false);
        acc10 = __builtin_amdgcn_wmma_f32_16x16x32_f16(false, a1, false, b0, (short)0, acc10, false, false);
        acc11 = __builtin_amdgcn_wmma_f32_16x16x32_f16(false, a1, false, b1, (short)0, acc11, false, false);
        __syncthreads();                               // done reading `cur` before reuse
    }

    store_tile(C, bias, R, acc00, m0,      n0,      M, N, act);
    store_tile(C, bias, R, acc01, m0,      n0 + 16, M, N, act);
    store_tile(C, bias, R, acc10, m0 + 16, n0,      M, N, act);
    store_tile(C, bias, R, acc11, m0 + 16, n0 + 16, M, N, act);
}

// ------------------------------- LayerNorm ---------------------------------
// One 256-thread block per row of 512.
__global__ __launch_bounds__(256) void layernorm_kernel(
    const float* __restrict__ X, const float* __restrict__ g,
    const float* __restrict__ b, float* __restrict__ O)
{
    __shared__ float red[256];
    const size_t row = blockIdx.x;
    const int t = threadIdx.x;
    const float* x = X + row * DM;
    float v0 = x[t], v1 = x[t + 256];

    red[t] = v0 + v1; __syncthreads();
    for (int o = 128; o > 0; o >>= 1) { if (t < o) red[t] += red[t + o]; __syncthreads(); }
    float mean = red[0] * (1.f / DM); __syncthreads();

    float d0 = v0 - mean, d1 = v1 - mean;
    red[t] = d0 * d0 + d1 * d1; __syncthreads();
    for (int o = 128; o > 0; o >>= 1) { if (t < o) red[t] += red[t + o]; __syncthreads(); }
    float inv = rsqrtf(red[0] * (1.f / DM) + 1e-5f);

    O[row * DM + t]       = d0 * inv * g[t]       + b[t];
    O[row * DM + t + 256] = d1 * inv * g[t + 256] + b[t + 256];
}

// ------------------------------- Embedding ---------------------------------
// circular conv (7->512, k=3) + sinusoidal positional embedding
__global__ void embed_kernel(const float* __restrict__ x, const float* __restrict__ cw,
                             float* __restrict__ out, int Lx)
{
    int t = blockIdx.x * 256 + threadIdx.x;
    int total = NB * Lx * DM;
    if (t >= total) return;
    int d = t & (DM - 1); int bl = t >> 9; int l = bl % Lx; int b = bl / Lx;
    float acc = 0.f;
#pragma unroll
    for (int tt = 0; tt < 3; ++tt) {
        int ls = l - 1 + tt; ls = (ls + Lx) % Lx;
        const float* xp = x + (size_t)(b * Lx + ls) * 7;
#pragma unroll
        for (int c = 0; c < 7; ++c) acc += xp[c] * cw[(d * 7 + c) * 3 + tt];
    }
    int   i2   = d >> 1;
    float freq = __expf(-(float)(2 * i2) * 0.0179889482f);  // ln(1e4)/512
    float ang  = (float)l * freq;
    float pe   = (d & 1) ? __cosf(ang) : __sinf(ang);
    out[t] = acc + pe;
}

// ------------------------- ProbSparse attention bits -----------------------

__global__ void gen_idx_kernel(int* __restrict__ idx, int LQ, int U, int LK, unsigned seed)
{
    int t = blockIdx.x * 256 + threadIdx.x;
    if (t >= LQ * U) return;
    unsigned h = (unsigned)t * 2654435761u + seed * 0x9E3779B9u + 0x7F4A7C15u;
    h ^= h >> 16; h *= 0x85EBCA6Bu; h ^= h >> 13; h *= 0xC2B2AE35u; h ^= h >> 16;
    idx[t] = (int)(h % (unsigned)LK);
}

// one wave per (b,h,lq): M = max_s(q.k_s) - mean over full L_K
__global__ __launch_bounds__(256) void sample_m_kernel(
    const float* __restrict__ Q, const float* __restrict__ Kb,
    const int* __restrict__ idx, float* __restrict__ Mout,
    int LQ, int LK, int U)
{
    int wave = blockIdx.x * 8 + (threadIdx.x >> 5);
    int lane = threadIdx.x & 31;
    int total = NB * NH * LQ;
    if (wave >= total) return;
    int lq = wave % LQ; int bh = wave / LQ; int h = bh % NH; int b = bh / NH;
    const float* q = Q + (size_t)(b * LQ + lq) * DM + h * DH;
    float mx = -INFINITY, sm = 0.f;
    for (int s = lane; s < U; s += 32) {
        int ki = idx[lq * U + s];
        const float* kp = Kb + (size_t)(b * LK + ki) * DM + h * DH;
        float d = 0.f;
#pragma unroll 8
        for (int e = 0; e < DH; ++e) d += q[e] * kp[e];
        mx = fmaxf(mx, d); sm += d;
    }
    for (int o = 16; o > 0; o >>= 1) {
        mx = fmaxf(mx, __shfl_xor(mx, o, 32));
        sm += __shfl_xor(sm, o, 32);
    }
    if (lane == 0) Mout[wave] = mx - sm / (float)LK;
}

// one block per (b,h): iterative top-u argmax over LQ (u <= 40)
__global__ __launch_bounds__(256) void topk_kernel(
    const float* __restrict__ Min, int* __restrict__ Mtop, int LQ, int u)
{
    __shared__ float vals[2048];
    __shared__ float rv[256];
    __shared__ int   ri[256];
    const float* m = Min + (size_t)blockIdx.x * LQ;
    for (int i = threadIdx.x; i < LQ; i += 256) vals[i] = m[i];
    __syncthreads();
    for (int it = 0; it < u; ++it) {
        float bv = -INFINITY; int bi = 0;
        for (int i = threadIdx.x; i < LQ; i += 256)
            if (vals[i] > bv) { bv = vals[i]; bi = i; }
        rv[threadIdx.x] = bv; ri[threadIdx.x] = bi; __syncthreads();
        for (int o = 128; o > 0; o >>= 1) {
            if (threadIdx.x < o && rv[threadIdx.x + o] > rv[threadIdx.x]) {
                rv[threadIdx.x] = rv[threadIdx.x + o];
                ri[threadIdx.x] = ri[threadIdx.x + o];
            }
            __syncthreads();
        }
        if (threadIdx.x == 0) { Mtop[blockIdx.x * u + it] = ri[0]; vals[ri[0]] = -INFINITY; }
        __syncthreads();
    }
}

// one block per (b,h,iu): scores -> softmax -> upd = attn @ V
__global__ __launch_bounds__(256) void attn_apply_kernel(
    const float* __restrict__ Q, const float* __restrict__ Kb,
    const float* __restrict__ V, const int* __restrict__ Mtop,
    float* __restrict__ Upd, int LQ, int LK, int u, int causal)
{
    __shared__ float sc[2048];
    __shared__ float red[256];
    int iu = blockIdx.x % u; int bh = blockIdx.x / u; int h = bh % NH; int b = bh / NH;
    int mq = Mtop[bh * u + iu];
    const float* q = Q + (size_t)(b * LQ + mq) * DM + h * DH;

    for (int k0 = threadIdx.x; k0 < LK; k0 += 256) {
        const float* kp = Kb + (size_t)(b * LK + k0) * DM + h * DH;
        float d = 0.f;
#pragma unroll 8
        for (int e = 0; e < DH; ++e) d += q[e] * kp[e];
        d *= 0.125f;                                  // 1/sqrt(64)
        if (causal && k0 > mq) d = -INFINITY;
        sc[k0] = d;
    }
    __syncthreads();

    float lm = -INFINITY;
    for (int k0 = threadIdx.x; k0 < LK; k0 += 256) lm = fmaxf(lm, sc[k0]);
    red[threadIdx.x] = lm; __syncthreads();
    for (int o = 128; o > 0; o >>= 1) {
        if (threadIdx.x < o) red[threadIdx.x] = fmaxf(red[threadIdx.x], red[threadIdx.x + o]);
        __syncthreads();
    }
    float mx = red[0]; __syncthreads();

    float ls = 0.f;
    for (int k0 = threadIdx.x; k0 < LK; k0 += 256) {
        float e = __expf(sc[k0] - mx); sc[k0] = e; ls += e;
    }
    red[threadIdx.x] = ls; __syncthreads();
    for (int o = 128; o > 0; o >>= 1) {
        if (threadIdx.x < o) red[threadIdx.x] += red[threadIdx.x + o];
        __syncthreads();
    }
    float inv = 1.f / red[0];

    if (threadIdx.x < DH) {
        int d = threadIdx.x; float s = 0.f;
        for (int k0 = 0; k0 < LK; ++k0)
            s += sc[k0] * V[(size_t)(b * LK + k0) * DM + h * DH + d];
        Upd[(size_t)blockIdx.x * DH + d] = s * inv;
    }
}

__global__ void meanv_kernel(const float* __restrict__ V, float* __restrict__ mean, int LK)
{
    int t = blockIdx.x * 256 + threadIdx.x;
    if (t >= NB * DM) return;
    int b = t / DM; int c = t % DM;
    float s = 0.f;
    for (int k = 0; k < LK; ++k) s += V[(size_t)(b * LK + k) * DM + c];
    mean[t] = s / (float)LK;
}

__global__ void ctxfill_kernel(const float* __restrict__ mean, float* __restrict__ ctx, int LQ)
{
    int t = blockIdx.x * 256 + threadIdx.x;
    if (t >= NB * LQ * DM) return;
    int c = t % DM; int b = (t / DM) / LQ;
    ctx[t] = mean[b * DM + c];
}

__global__ void cumsum_kernel(const float* __restrict__ V, float* __restrict__ ctx, int Lx)
{
    int t = blockIdx.x * 256 + threadIdx.x;
    if (t >= NB * DM) return;
    int b = t / DM; int c = t % DM;
    float s = 0.f;
    for (int lq = 0; lq < Lx; ++lq) {
        s += V[(size_t)(b * Lx + lq) * DM + c];
        ctx[(size_t)(b * Lx + lq) * DM + c] = s;
    }
}

__global__ void scatter_kernel(const float* __restrict__ Upd, const int* __restrict__ Mtop,
                               float* __restrict__ ctx, int LQ, int u)
{
    int t = blockIdx.x * 256 + threadIdx.x;
    if (t >= NB * NH * u * DH) return;
    int d = t % DH; int rest = t / DH; int iu = rest % u; int bh = rest / u;
    int b = bh / NH; int h = bh % NH;
    int row = b * LQ + Mtop[bh * u + iu];
    ctx[(size_t)row * DM + h * DH + d] = Upd[(size_t)(bh * u + iu) * DH + d];
}

// ------------------------------- Distil conv -------------------------------

__global__ void im2col_kernel(const float* __restrict__ x, float* __restrict__ col, int Lx)
{
    int t = blockIdx.x * 256 + threadIdx.x;
    if (t >= NB * Lx * KCOL) return;
    int k = t % KCOL; int bl = t / KCOL; int l = bl % Lx; int b = bl / Lx;
    int tt = k / DM; int c = k % DM;
    int ls = (l - 1 + tt + Lx) % Lx;
    col[t] = x[(size_t)(b * Lx + ls) * DM + c];
}

__global__ void repackw_kernel(const float* __restrict__ wsrc, float* __restrict__ wcol)
{
    int t = blockIdx.x * 256 + threadIdx.x;
    if (t >= DM * KCOL) return;
    int dout = t / KCOL; int k = t % KCOL; int tt = k / DM; int c = k % DM;
    wcol[t] = wsrc[((size_t)dout * DM + c) * 3 + tt];
}

// maxpool k=3 stride=2 on -inf padded sequence (length Lx -> Lx/2)
__global__ void maxpool_kernel(const float* __restrict__ y, float* __restrict__ o, int Lx)
{
    int Lo = Lx / 2;
    int t = blockIdx.x * 256 + threadIdx.x;
    if (t >= NB * Lo * DM) return;
    int d = t % DM; int bl = t / DM; int lp = bl % Lo; int b = bl / Lo;
    float m = -INFINITY;
#pragma unroll
    for (int w = -1; w <= 1; ++w) {
        int l = 2 * lp + w;
        if (l >= 0 && l < Lx) m = fmaxf(m, y[(size_t)(b * Lx + l) * DM + d]);
    }
    o[t] = m;
}

// --------------------------- tiny output MLP -------------------------------
// per row: relu(x@r1^T+b1) -> relu(@r2^T+b2) -> @r3^T+b3
__global__ void final_mlp_kernel(const float* __restrict__ X,
                                 const float* __restrict__ r1w, const float* __restrict__ r1b,
                                 const float* __restrict__ r2w, const float* __restrict__ r2b,
                                 const float* __restrict__ r3w, const float* __restrict__ r3b,
                                 float* __restrict__ out, int rows)
{
    int r = blockIdx.x * 256 + threadIdx.x;
    if (r >= rows) return;
    float a[24], h1[24], h2[12];
#pragma unroll
    for (int i = 0; i < 24; ++i) a[i] = X[(size_t)r * 24 + i];
#pragma unroll
    for (int o = 0; o < 24; ++o) {
        float s = r1b[o];
        for (int i = 0; i < 24; ++i) s += r1w[o * 24 + i] * a[i];
        h1[o] = fmaxf(s, 0.f);
    }
#pragma unroll
    for (int o = 0; o < 12; ++o) {
        float s = r2b[o];
        for (int i = 0; i < 24; ++i) s += r2w[o * 24 + i] * h1[i];
        h2[o] = fmaxf(s, 0.f);
    }
    float s = r3b[0];
#pragma unroll
    for (int i = 0; i < 12; ++i) s += r3w[i] * h2[i];
    out[r] = s;
}

// ------------------------------- host side ---------------------------------

struct AttnP { const float *wq,*bq,*wk,*bk,*wv,*bv,*wo,*bo; };
struct EncP  { AttnP a; const float *ln1g,*ln1b,*ln2g,*ln2b,*w1,*b1,*w2,*b2; };
struct DecP  { AttnP self, cross;
               const float *ln1g,*ln1b,*ln2g,*ln2b,*ln3g,*ln3b,*w1,*b1,*w2,*b2; };

static inline unsigned cdiv(size_t a, unsigned b) { return (unsigned)((a + b - 1) / b); }
static inline int probe_u(int Lc) {
    int v = (int)(5.0 * std::ceil(std::log((double)Lc)));
    return v < Lc ? v : Lc;
}

extern "C" void kernel_launch(void* const* d_in, const int* in_sizes, int n_in,
                              void* d_out, int out_size, void* d_ws, size_t ws_size,
                              hipStream_t stream)
{
    (void)in_sizes; (void)n_in; (void)out_size; (void)ws_size;

    // ---- parse params in setup_inputs() insertion order ----
    int ip = 0;
    auto nf = [&]() { return (const float*)d_in[ip++]; };
    auto attn_parse = [&](AttnP& a) {
        a.wq = nf(); a.bq = nf(); a.wk = nf(); a.bk = nf();
        a.wv = nf(); a.bv = nf(); a.wo = nf(); a.bo = nf();
    };
    const float* enc_conv = nf();
    const float* dec_conv = nf();
    EncP enc[2];
    for (int i = 0; i < 2; ++i) {
        attn_parse(enc[i].a);
        enc[i].ln1g = nf(); enc[i].ln1b = nf(); enc[i].ln2g = nf(); enc[i].ln2b = nf();
        enc[i].w1 = nf(); enc[i].b1 = nf(); enc[i].w2 = nf(); enc[i].b2 = nf();
    }
    const float* distil_w = nf(); const float* distil_b = nf();
    const float* enc_lng = nf();  const float* enc_lnb = nf();
    DecP dec;
    attn_parse(dec.self);
    attn_parse(dec.cross);
    dec.ln1g = nf(); dec.ln1b = nf(); dec.ln2g = nf(); dec.ln2b = nf();
    dec.ln3g = nf(); dec.ln3b = nf();
    dec.w1 = nf(); dec.b1 = nf(); dec.w2 = nf(); dec.b2 = nf();
    const float* dec_lng = nf(); const float* dec_lnb = nf();
    const float* proj_w = nf();  const float* proj_b = nf();
    const float* r1w = nf(); const float* r1b = nf();
    const float* r2w = nf(); const float* r2b = nf();
    const float* r3w = nf(); const float* r3b = nf();
    const float* x_enc = nf();

    // ---- carve workspace ----
    const size_t RMAX = (size_t)NB * SL;   // 8192 rows max
    float* wsb = (float*)d_ws;
    size_t off = 0;
    auto alloc = [&](size_t n) { float* p = wsb + off; off += n; return p; };
    float* Bx    = alloc(RMAX * DM);
    float* By    = alloc(RMAX * DM);
    float* Bq    = alloc(RMAX * DM);
    float* Bk    = alloc(RMAX * DM);
    float* Bv    = alloc(RMAX * DM);
    float* Bctx  = alloc(RMAX * DM);
    float* Bcol  = alloc(RMAX * KCOL);     // also reused as FFN scratch (RMAX*DFF fits)
    float* Bff   = Bcol;
    float* Bwcol = alloc((size_t)DM * KCOL);
    float* Bm    = alloc((size_t)NB * NH * SL);
    float* Bupd  = alloc((size_t)NB * NH * 40 * DH);
    float* Bmean = alloc((size_t)NB * DM);
    float* Bproj = alloc(RMAX * 24);
    int*   Bidx  = (int*)alloc((size_t)SL * 40);
    int*   Btop  = (int*)alloc((size_t)NB * NH * 40);

    // ---- launch helpers ----
    auto gemm = [&](const float* A, const float* Wt, const float* bias, const float* R,
                    float* Cp, int M, int N, int K, int act) {
        dim3 g((unsigned)((N + 63) / 64), (unsigned)((M + 127) / 128));
        gemm_wmma_kernel<<<g, 256, 0, stream>>>(A, Wt, bias, R, Cp, M, N, K, act);
    };
    auto lnorm = [&](const float* X, const float* g_, const float* b_, float* O, int rows) {
        layernorm_kernel<<<(unsigned)rows, 256, 0, stream>>>(X, g_, b_, O);
    };
    auto attention = [&](const float* x, const float* cross, int LQ, int LK,
                         int U, int u, unsigned seed, int causal,
                         const AttnP& P, float* out) {
        int rq = NB * LQ, rk = NB * LK;
        gemm(x,     P.wq, P.bq, nullptr, Bq, rq, DM, DM, ACT_NONE);
        gemm(cross, P.wk, P.bk, nullptr, Bk, rk, DM, DM, ACT_NONE);
        gemm(cross, P.wv, P.bv, nullptr, Bv, rk, DM, DM, ACT_NONE);
        gen_idx_kernel<<<cdiv((size_t)LQ * U, 256), 256, 0, stream>>>(Bidx, LQ, U, LK, seed);
        int waves = NB * NH * LQ;
        sample_m_kernel<<<cdiv(waves, 8), 256, 0, stream>>>(Bq, Bk, Bidx, Bm, LQ, LK, U);
        topk_kernel<<<NB * NH, 256, 0, stream>>>(Bm, Btop, LQ, u);
        attn_apply_kernel<<<NB * NH * u, 256, 0, stream>>>(Bq, Bk, Bv, Btop, Bupd,
                                                           LQ, LK, u, causal);
        if (causal) {
            cumsum_kernel<<<cdiv((size_t)NB * DM, 256), 256, 0, stream>>>(Bv, Bctx, LQ);
        } else {
            meanv_kernel<<<cdiv((size_t)NB * DM, 256), 256, 0, stream>>>(Bv, Bmean, LK);
            ctxfill_kernel<<<cdiv((size_t)rq * DM, 256), 256, 0, stream>>>(Bmean, Bctx, LQ);
        }
        scatter_kernel<<<cdiv((size_t)NB * NH * u * DH, 256), 256, 0, stream>>>(Bupd, Btop,
                                                                                Bctx, LQ, u);
        gemm(Bctx, P.wo, P.bo, /*residual=*/x, out, rq, DM, DM, ACT_NONE);  // x + attn(x)
    };
    auto enc_layer = [&](float* x, int Lx, const EncP& p, unsigned seed) {
        int uu = probe_u(Lx);
        int rows = NB * Lx;
        attention(x, x, Lx, Lx, uu, uu, seed, 0, p.a, By);
        lnorm(By, p.ln1g, p.ln1b, x, rows);
        gemm(x, p.w1, p.b1, nullptr, Bff, rows, DFF, DM, ACT_GELU);
        gemm(Bff, p.w2, p.b2, /*residual=*/x, By, rows, DM, DFF, ACT_NONE);
        lnorm(By, p.ln2g, p.ln2b, x, rows);
    };

    // ---------------- encoder ----------------
    embed_kernel<<<cdiv((size_t)NB * SL * DM, 256), 256, 0, stream>>>(x_enc, enc_conv, Bx, SL);
    enc_layer(Bx, SL, enc[0], 0);

    im2col_kernel<<<cdiv((size_t)NB * SL * KCOL, 256), 256, 0, stream>>>(Bx, Bcol, SL);
    repackw_kernel<<<cdiv((size_t)DM * KCOL, 256), 256, 0, stream>>>(distil_w, Bwcol);
    gemm(Bcol, Bwcol, distil_b, nullptr, By, NB * SL, DM, KCOL, ACT_ELU);
    maxpool_kernel<<<cdiv((size_t)NB * (SL / 2) * DM, 256), 256, 0, stream>>>(By, Bx, SL);

    enc_layer(Bx, SL / 2, enc[1], 10);
    float* enc_out = (float*)d_out;                       // [4,1024,512] first output
    lnorm(Bx, enc_lng, enc_lnb, enc_out, NB * (SL / 2));

    // ---------------- decoder ----------------
    embed_kernel<<<cdiv((size_t)NB * SL * DM, 256), 256, 0, stream>>>(x_enc, dec_conv, Bx, SL);
    attention(Bx, Bx, SL, SL, probe_u(SL), probe_u(SL), 20, 1, dec.self, By);
    lnorm(By, dec.ln1g, dec.ln1b, Bx, NB * SL);
    attention(Bx, enc_out, SL, SL / 2, probe_u(SL / 2), probe_u(SL), 21, 0, dec.cross, By);
    lnorm(By, dec.ln2g, dec.ln2b, Bx, NB * SL);
    gemm(Bx, dec.w1, dec.b1, nullptr, Bff, NB * SL, DFF, DM, ACT_GELU);
    gemm(Bff, dec.w2, dec.b2, /*residual=*/Bx, By, NB * SL, DM, DFF, ACT_NONE);
    lnorm(By, dec.ln3g, dec.ln3b, Bx, NB * SL);
    lnorm(Bx, dec_lng, dec_lnb, By, NB * SL);

    gemm(By, proj_w, proj_b, nullptr, Bproj, NB * SL, 24, DM, ACT_NONE);
    final_mlp_kernel<<<cdiv((size_t)NB * SL, 256), 256, 0, stream>>>(
        Bproj, r1w, r1b, r2w, r2b, r3w, r3b,
        enc_out + (size_t)NB * (SL / 2) * DM, NB * SL);   // second output [4,2048,1]
}
